// TransformerEncoderLayer_5463198401292
// MI455X (gfx1250) — matrix operand order repaired and checked by reading, not compiled, not run on hardware
//
#include <hip/hip_runtime.h>
#include <math.h>

#define BB 4
#define SS 2048
#define EE 1024
#define HH 16
#define DD 64
#define FFN 4096
#define RR (BB * SS) /* 8192 tokens */

typedef __attribute__((ext_vector_type(16))) __bf16 v16bf;
typedef __attribute__((ext_vector_type(8)))  __bf16 v8bf;
typedef __attribute__((ext_vector_type(8)))  float  v8f;
typedef __attribute__((ext_vector_type(4)))  unsigned u32x4;
typedef __attribute__((ext_vector_type(8)))  int      i32x8;
typedef __attribute__((ext_vector_type(4)))  int      i32x4;

// ---------- helpers ----------
static __device__ __forceinline__ __bf16 f2bf(float f) {
  return (__bf16)f;  // native cvt on gfx1250
}

static __device__ __forceinline__ v8f vzero() {
  v8f z;
#pragma unroll
  for (int i = 0; i < 8; ++i) z[i] = 0.0f;
  return z;
}

static __device__ __forceinline__ v8f wmma_bf16(v16bf a, v16bf b, v8f c) {
  // D = A(16x32 bf16) * B(32x16 bf16) + C(16x16 f32)
  return __builtin_amdgcn_wmma_f32_16x16x32_bf16(
      /*neg_a=*/false, a, /*neg_b=*/false, b,
      /*c_mod=*/(short)0, c, /*reuse_a=*/false, /*reuse_b=*/false);
}

// A-operand (16x32, bf16) loader from row-major [M][K] memory (ld in elems).
// ISA layout: lane<16 holds M=lane, K={0..7,16..23}; lane>=16: K={8..15,24..31}.
static __device__ __forceinline__ v16bf load_a_tile(const __bf16* base, int ld, int lane) {
  int m = lane & 15;
  int ks = (lane >> 4) << 3;  // 0 or 8
  const __bf16* p = base + (size_t)m * ld + ks;
  v8bf lo = *(const v8bf*)(p);
  v8bf hi = *(const v8bf*)(p + 16);
  v16bf r;
#pragma unroll
  for (int i = 0; i < 8; ++i) { r[i] = lo[i]; r[i + 8] = hi[i]; }
  return r;
}

// B-operand (32x16, bf16) loader from row-major [N][K] memory (ld in elems),
// base points at element [n0][k0]. Lane n holds column n0+n, K={0..7,16..23} (lane<16) or {8..15,24..31}.
static __device__ __forceinline__ v16bf load_b_nk(const __bf16* base, int ld, int lane) {
  int n = lane & 15;
  int ks = (lane >> 4) << 3;  // 0 or 8
  const __bf16* p = base + (size_t)n * ld + ks;
  v8bf lo = *(const v8bf*)(p);
  v8bf hi = *(const v8bf*)(p + 16);
  v16bf r;
#pragma unroll
  for (int i = 0; i < 8; ++i) { r[i] = lo[i]; r[i + 8] = hi[i]; }
  return r;
}

// ---------- Tensor Data Mover: 2-D tile (rows x cols bf16) global -> LDS ----------
// D# per CDNA5 ISA: group0 = {flags, lds_addr, global_addr,
// type=2}; group1 packs data_size=1(2B), tensor_dim0=row_stride, tensor_dim1=rows,
// tile_dim0=cols, tile_dim1=rows, tensor_dim0_stride=row_stride. 2-D: groups 2/3 zero.
static __device__ __forceinline__ void tdm_load_2d(const __bf16* gsrc, void* lds_dst,
                                                   int rows, int cols_elems,
                                                   int row_stride_elems) {
  unsigned long long ga = (unsigned long long)(size_t)gsrc;
  u32x4 g0;
  g0[0] = 1u;                                                 // count=1
  g0[1] = (unsigned)(size_t)lds_dst;                          // lds_addr
  g0[2] = (unsigned)(ga & 0xffffffffu);                       // global_addr lo
  g0[3] = (unsigned)((ga >> 32) & 0x01ffffffu) | (2u << 30);  // addr hi | type=2
  unsigned k = (unsigned)row_stride_elems;
  unsigned rr = (unsigned)rows, cc = (unsigned)cols_elems;
  i32x8 g1;
  g1[0] = (int)(1u << 16);                                    // data_size=1 (2B)
  g1[1] = (int)((k & 0xffffu) << 16);                         // tensor_dim0 lo16
  g1[2] = (int)(((k >> 16) & 0xffffu) | ((rr & 0xffffu) << 16));  // dim0 hi|dim1 lo
  g1[3] = (int)((cc & 0xffffu) << 16);                        // tile_dim0
  g1[4] = (int)(rr & 0xffffu);                                // tile_dim1
  g1[5] = (int)k;                                             // dim0_stride lo32
  g1[6] = 0;
  g1[7] = 0;
  i32x4 z4 = {0, 0, 0, 0};
#if defined(__clang_major__) && (__clang_major__ >= 23)
  i32x8 z8 = {0, 0, 0, 0, 0, 0, 0, 0};
  __builtin_amdgcn_tensor_load_to_lds(g0, g1, z4, z4, z8, 0);
#else
  __builtin_amdgcn_tensor_load_to_lds(g0, g1, z4, z4, 0);
#endif
}

// ---------- kernel 0: fp32 -> bf16 conversion ----------
__global__ void cvt_bf16_kernel(const float* __restrict__ in, __bf16* __restrict__ out, int n) {
  int i = blockIdx.x * blockDim.x + threadIdx.x;
  if (i < n) out[i] = f2bf(in[i]);
}

// ---------- kernel 1: LayerNorm (row of 1024) -> bf16 ----------
__global__ void ln_bf16_kernel(const float* __restrict__ x, const float* __restrict__ g,
                               const float* __restrict__ b, __bf16* __restrict__ out) {
  int row = blockIdx.x;
  const float* xr = x + (size_t)row * EE;
  float v[4];
  float s = 0.f, s2 = 0.f;
#pragma unroll
  for (int i = 0; i < 4; ++i) {
    v[i] = xr[threadIdx.x + 256 * i];
    s += v[i];
    s2 += v[i] * v[i];
  }
#pragma unroll
  for (int m = 16; m >= 1; m >>= 1) {
    s += __shfl_xor(s, m, 32);
    s2 += __shfl_xor(s2, m, 32);
  }
  __shared__ float ps[8], ps2[8];
  int wave = threadIdx.x >> 5, lane = threadIdx.x & 31;
  if (lane == 0) { ps[wave] = s; ps2[wave] = s2; }
  __syncthreads();
  float S1 = 0.f, S2 = 0.f;
#pragma unroll
  for (int i = 0; i < 8; ++i) { S1 += ps[i]; S2 += ps2[i]; }
  float mu = S1 * (1.0f / EE);
  float var = S2 * (1.0f / EE) - mu * mu;
  float inv = rsqrtf(var + 1e-5f);
#pragma unroll
  for (int i = 0; i < 4; ++i) {
    int c = threadIdx.x + 256 * i;
    out[(size_t)row * EE + c] = f2bf((v[i] - mu) * inv * g[c] + b[c]);
  }
}

// ---------- kernel 2: per-head projection, X(131072x64) @ W(64x64)^T ----------
// Rows are (token*H + head); row data is contiguous (since E = H*D).
// Whole 64x64 weight staged in LDS once per block via TDM; 8 waves reuse it.
__global__ void head_proj_kernel(const __bf16* __restrict__ X, const __bf16* __restrict__ W,
                                 __bf16* __restrict__ outN, __bf16* __restrict__ outT,
                                 int transpose) {
  __shared__ __bf16 wt[64][64];  // 8 KB
  int lane = threadIdx.x & 31;
  int wave = threadIdx.x >> 5;
  if (wave == 0) {
    tdm_load_2d(W, &wt[0][0], 64, 64, 64);
    __builtin_amdgcn_s_wait_tensorcnt(0);
  }
  __syncthreads();

  int wid = blockIdx.x * (blockDim.x >> 5) + wave;
  size_t row0 = (size_t)wid * 16;
  const __bf16* abase = X + row0 * DD;
  v16bf a0 = load_a_tile(abase, DD, lane);
  v16bf a1 = load_a_tile(abase + 32, DD, lane);
  int n = lane & 15, m0 = (lane >> 4) << 3;
#pragma unroll
  for (int nt = 0; nt < 4; ++nt) {
    v8f c = vzero();
    v16bf b0 = load_b_nk(&wt[nt * 16][0], DD, lane);
    v16bf b1 = load_b_nk(&wt[nt * 16][32], DD, lane);
    c = wmma_bf16(a0, b0, c);
    c = wmma_bf16(a1, b1, c);
    if (!transpose) {
#pragma unroll
      for (int r = 0; r < 8; ++r) {
        size_t rw = row0 + m0 + r;
        size_t tok = rw >> 4;
        int head = (int)(rw & 15);
        outN[tok * EE + head * DD + nt * 16 + n] = f2bf(c[r]);
      }
    } else {
      // V stored transposed: [b][h][d][s] so attention B-frags load contiguous
#pragma unroll
      for (int r = 0; r < 8; ++r) {
        size_t rw = row0 + m0 + r;
        size_t tok = rw >> 4;
        int head = (int)(rw & 15);
        size_t bi = tok / SS, si = tok % SS;
        int d = nt * 16 + n;
        outT[(((bi * HH) + head) * DD + d) * SS + si] = f2bf(c[r]);
      }
    }
  }
}

// ---------- kernel 3: flash attention, one 16-row q-tile per wave ----------
__global__ void flash_attn_kernel(const __bf16* __restrict__ Q, const __bf16* __restrict__ K,
                                  const __bf16* __restrict__ VT, const int* __restrict__ mask,
                                  __bf16* __restrict__ O) {
  __shared__ __bf16 lds_p[4][16][40];  // P tile staging, padded (80B rows, 16B aligned)
  int lane = threadIdx.x & 31;
  int wave = threadIdx.x >> 5;
  int bh = blockIdx.y;  // 0..63
  int batch = bh >> 4, head = bh & 15;
  int q0 = blockIdx.x * 64 + wave * 16;

  const __bf16* qp = Q + (size_t)batch * SS * EE + head * DD;
  const __bf16* kp = K + (size_t)batch * SS * EE + head * DD;
  const __bf16* vt = VT + (size_t)bh * DD * SS;  // [D][S] for this (b,h)
  const int* mrow = mask + (size_t)batch * SS;

  v16bf aq0 = load_a_tile(qp + (size_t)q0 * EE, EE, lane);
  v16bf aq1 = load_a_tile(qp + (size_t)q0 * EE + 32, EE, lane);

  float slope = exp2f(-(float)(head + 1));
  const float sc = 1.0f / 32.0f;  // 1/sqrt(E)

  v8f o0 = vzero(), o1 = vzero(), o2 = vzero(), o3 = vzero();
  float mrun[8], lrun[8];
#pragma unroll
  for (int r = 0; r < 8; ++r) { mrun[r] = -3.0e38f; lrun[r] = 0.f; }

  int n = lane & 15;
  int m0 = (lane >> 4) << 3;

  for (int kb = 0; kb < SS; kb += 32) {
    // ---- S = Q @ K^T for 32 keys (2 column tiles x 2 k-steps) ----
    v8f s0 = vzero(), s1 = vzero();
    {
      const __bf16* kbase = kp + (size_t)kb * EE;
      __builtin_prefetch(kbase + 32 * EE);
      v16bf b;
      b = load_b_nk(kbase, EE, lane);                s0 = wmma_bf16(aq0, b, s0);
      b = load_b_nk(kbase + 32, EE, lane);           s0 = wmma_bf16(aq1, b, s0);
      b = load_b_nk(kbase + 16 * EE, EE, lane);      s1 = wmma_bf16(aq0, b, s1);
      b = load_b_nk(kbase + 16 * EE + 32, EE, lane); s1 = wmma_bf16(aq1, b, s1);
    }
    int k0g = kb + n, k1g = kb + 16 + n;
    int msk0 = mrow[k0g], msk1 = mrow[k1g];
    float p0[8], p1[8], rsc[8];
#pragma unroll
    for (int r = 0; r < 8; ++r) {
      int qi = q0 + m0 + r;
      float e0 = s0[r] - slope * fabsf((float)(qi - k0g));
      float e1 = s1[r] - slope * fabsf((float)(qi - k1g));
      if (msk0 == 0) e0 = -1.0e4f;
      if (msk1 == 0) e1 = -1.0e4f;
      e0 *= sc; e1 *= sc;
      float mx = fmaxf(e0, e1);
#pragma unroll
      for (int d = 8; d >= 1; d >>= 1) mx = fmaxf(mx, __shfl_xor(mx, d, 32));
      float mnew = fmaxf(mrun[r], mx);
      float scale = __expf(mrun[r] - mnew);
      float q0e = __expf(e0 - mnew);
      float q1e = __expf(e1 - mnew);
      float rs = q0e + q1e;
#pragma unroll
      for (int d = 8; d >= 1; d >>= 1) rs += __shfl_xor(rs, d, 32);
      lrun[r] = lrun[r] * scale + rs;
      mrun[r] = mnew;
      p0[r] = q0e; p1[r] = q1e; rsc[r] = scale;
    }
    // ---- rescale O; stage P (C-layout -> LDS -> A-layout) ----
#pragma unroll
    for (int r = 0; r < 8; ++r) {
      o0[r] *= rsc[r]; o1[r] *= rsc[r]; o2[r] *= rsc[r]; o3[r] *= rsc[r];
      lds_p[wave][m0 + r][n] = f2bf(p0[r]);
      lds_p[wave][m0 + r][16 + n] = f2bf(p1[r]);
    }
    asm volatile("s_wait_dscnt 0" ::: "memory");  // same-wave DS ops are in order
    v16bf ap = load_a_tile(&lds_p[wave][0][0], 40, lane);
    // ---- O += P @ V ; VT is [D][S] row-major == [N][K], ld = S ----
    const __bf16* vbase = vt + kb;
    v16bf b;
    b = load_b_nk(vbase, SS, lane);           o0 = wmma_bf16(ap, b, o0);
    b = load_b_nk(vbase + 16 * SS, SS, lane); o1 = wmma_bf16(ap, b, o1);
    b = load_b_nk(vbase + 32 * SS, SS, lane); o2 = wmma_bf16(ap, b, o2);
    b = load_b_nk(vbase + 48 * SS, SS, lane); o3 = wmma_bf16(ap, b, o3);
  }
  // ---- normalize and store context (token-major, head-major within E) ----
  __bf16* ob = O + ((size_t)batch * SS + q0) * EE + head * DD;
#pragma unroll
  for (int r = 0; r < 8; ++r) {
    float inv = 1.0f / lrun[r];
    size_t ro = (size_t)(m0 + r) * EE;
    ob[ro + n]      = f2bf(o0[r] * inv);
    ob[ro + 16 + n] = f2bf(o1[r] * inv);
    ob[ro + 32 + n] = f2bf(o2[r] * inv);
    ob[ro + 48 + n] = f2bf(o3[r] * inv);
  }
}

// ---------- kernel 4: C = A @ W^T + bias + residual, fp32 out ----------
// Block = 8 waves covering 128 rows x 64 cols; W k-slab staged in LDS via TDM.
__global__ void gemm_bias_res_kernel(const __bf16* __restrict__ A, const __bf16* __restrict__ W,
                                     const float* __restrict__ bias, const float* __restrict__ res,
                                     float* __restrict__ out, int N, int Kd) {
  __shared__ __bf16 wt[64][32];  // 4 KB weight slab
  int lane = threadIdx.x & 31;
  int wave = threadIdx.x >> 5;
  int n0 = blockIdx.y << 6;
  size_t row0 = ((size_t)blockIdx.x * 8 + wave) << 4;
  v8f acc[4];
#pragma unroll
  for (int nt = 0; nt < 4; ++nt) acc[nt] = vzero();
  const __bf16* abase = A + row0 * Kd;
  for (int k = 0; k < Kd; k += 32) {
    __syncthreads();  // previous slab fully consumed
    if (wave == 0) {
      tdm_load_2d(W + (size_t)n0 * Kd + k, &wt[0][0], 64, 32, Kd);
      __builtin_amdgcn_s_wait_tensorcnt(0);
    }
    __syncthreads();
    __builtin_prefetch(abase + k + 128);
    v16bf a = load_a_tile(abase + k, Kd, lane);
#pragma unroll
    for (int nt = 0; nt < 4; ++nt) {
      v16bf b = load_b_nk(&wt[nt * 16][0], 32, lane);
      acc[nt] = wmma_bf16(a, b, acc[nt]);
    }
  }
  int n = lane & 15, m0 = (lane >> 4) << 3;
#pragma unroll
  for (int nt = 0; nt < 4; ++nt) {
#pragma unroll
    for (int r = 0; r < 8; ++r) {
      size_t rr = row0 + m0 + r;
      int cc = n0 + nt * 16 + n;
      float v = acc[nt][r] + bias[cc];
      v += res[rr * N + cc];
      out[rr * N + cc] = v;
    }
  }
}

// ---------- kernel 5: fused SwiGLU, Y = silu(A@W1^T + b1) * (A@W2^T + b2) ----------
__global__ void gemm_swiglu_kernel(const __bf16* __restrict__ A, const __bf16* __restrict__ W12,
                                   const float* __restrict__ b12, __bf16* __restrict__ Y) {
  __shared__ __bf16 w1t[64][32];  // 4 KB
  __shared__ __bf16 w2t[64][32];  // 4 KB
  int lane = threadIdx.x & 31;
  int wave = threadIdx.x >> 5;
  int n0 = blockIdx.y << 6;
  size_t row0 = ((size_t)blockIdx.x * 8 + wave) << 4;
  v8f acc1[4], acc2[4];
#pragma unroll
  for (int nt = 0; nt < 4; ++nt) { acc1[nt] = vzero(); acc2[nt] = vzero(); }
  const __bf16* abase = A + row0 * EE;
  for (int k = 0; k < EE; k += 32) {
    __syncthreads();
    if (wave == 0) {
      tdm_load_2d(W12 + (size_t)n0 * EE + k, &w1t[0][0], 64, 32, EE);
      tdm_load_2d(W12 + (size_t)(FFN + n0) * EE + k, &w2t[0][0], 64, 32, EE);
      __builtin_amdgcn_s_wait_tensorcnt(0);
    }
    __syncthreads();
    __builtin_prefetch(abase + k + 128);
    v16bf a = load_a_tile(abase + k, EE, lane);
#pragma unroll
    for (int nt = 0; nt < 4; ++nt) {
      v16bf b1 = load_b_nk(&w1t[nt * 16][0], 32, lane);
      acc1[nt] = wmma_bf16(a, b1, acc1[nt]);
      v16bf b2 = load_b_nk(&w2t[nt * 16][0], 32, lane);
      acc2[nt] = wmma_bf16(a, b2, acc2[nt]);
    }
  }
  int n = lane & 15, m0 = (lane >> 4) << 3;
#pragma unroll
  for (int nt = 0; nt < 4; ++nt) {
#pragma unroll
    for (int r = 0; r < 8; ++r) {
      int cc = n0 + nt * 16 + n;
      float x1 = acc1[nt][r] + b12[cc];
      float x2 = acc2[nt][r] + b12[FFN + cc];
      float sil = x1 / (1.0f + __expf(-x1));
      Y[(row0 + m0 + r) * (size_t)FFN + cc] = f2bf(sil * x2);
    }
  }
}

// ---------- host launcher ----------
extern "C" void kernel_launch(void* const* d_in, const int* in_sizes, int n_in,
                              void* d_out, int out_size, void* d_ws, size_t ws_size,
                              hipStream_t stream) {
  (void)in_sizes; (void)n_in; (void)out_size; (void)ws_size;
  const float* d_value = (const float*)d_in[0];
  const float* d_key   = (const float*)d_in[1];
  const float* d_query = (const float*)d_in[2];
  const int*   d_mask  = (const int*)d_in[3];
  const float* g1 = (const float*)d_in[4];
  const float* b1 = (const float*)d_in[5];
  const float* g2 = (const float*)d_in[6];
  const float* b2 = (const float*)d_in[7];
  const float* Wv = (const float*)d_in[8];
  const float* Wk = (const float*)d_in[9];
  const float* Wq = (const float*)d_in[10];
  const float* Wo = (const float*)d_in[11];
  const float* bo = (const float*)d_in[12];
  const float* W12 = (const float*)d_in[13];
  const float* b12 = (const float*)d_in[14];
  const float* W3 = (const float*)d_in[15];
  const float* b3 = (const float*)d_in[16];
  float* out = (float*)d_out;

  size_t off = 0;
  char* wsb = (char*)d_ws;
  auto alloc = [&](size_t bytes) -> void* {
    void* p = wsb + off;
    off += (bytes + 255) & ~(size_t)255;
    return p;
  };
  const size_t tokE = (size_t)RR * EE;
  __bf16* qn  = (__bf16*)alloc(tokE * 2);
  __bf16* kn  = (__bf16*)alloc(tokE * 2);
  __bf16* vn  = (__bf16*)alloc(tokE * 2);
  __bf16* qp  = (__bf16*)alloc(tokE * 2);
  __bf16* kp  = (__bf16*)alloc(tokE * 2);
  __bf16* vT  = (__bf16*)alloc(tokE * 2);
  __bf16* ctx = (__bf16*)alloc(tokE * 2);
  float*  hres = (float*)alloc(tokE * 4);
  __bf16* x2n = (__bf16*)alloc(tokE * 2);
  __bf16* y   = (__bf16*)alloc((size_t)RR * FFN * 2);
  __bf16* wq_b = (__bf16*)alloc((size_t)DD * DD * 2);
  __bf16* wk_b = (__bf16*)alloc((size_t)DD * DD * 2);
  __bf16* wv_b = (__bf16*)alloc((size_t)DD * DD * 2);
  __bf16* wo_b = (__bf16*)alloc((size_t)EE * EE * 2);
  __bf16* w12_b = (__bf16*)alloc((size_t)2 * FFN * EE * 2);
  __bf16* w3_b = (__bf16*)alloc((size_t)EE * FFN * 2);

  // 0) weights -> bf16
  auto cvt = [&](const float* src, __bf16* dst, int n) {
    cvt_bf16_kernel<<<(n + 255) / 256, 256, 0, stream>>>(src, dst, n);
  };
  cvt(Wq, wq_b, DD * DD);
  cvt(Wk, wk_b, DD * DD);
  cvt(Wv, wv_b, DD * DD);
  cvt(Wo, wo_b, EE * EE);
  cvt(W12, w12_b, 2 * FFN * EE);
  cvt(W3, w3_b, EE * FFN);

  // 1) LayerNorm(q,k,v) -> bf16
  ln_bf16_kernel<<<RR, 256, 0, stream>>>(d_query, g1, b1, qn);
  ln_bf16_kernel<<<RR, 256, 0, stream>>>(d_key,   g1, b1, kn);
  ln_bf16_kernel<<<RR, 256, 0, stream>>>(d_value, g1, b1, vn);

  // 2) per-head projections (V transposed for attention B-operand)
  {
    int blocks = (RR * HH / 16) / 8;  // 1024 blocks of 8 waves
    head_proj_kernel<<<blocks, 256, 0, stream>>>(qn, wq_b, qp, nullptr, 0);
    head_proj_kernel<<<blocks, 256, 0, stream>>>(kn, wk_b, kp, nullptr, 0);
    head_proj_kernel<<<blocks, 256, 0, stream>>>(vn, wv_b, nullptr, vT, 1);
  }

  // 3) flash attention
  flash_attn_kernel<<<dim3(SS / 64, BB * HH), 128, 0, stream>>>(qp, kp, vT, d_mask, ctx);

  // 4) Wo GEMM + bo + residual(query) -> fp32 h   (grid: 128-row groups x 64-col slabs)
  gemm_bias_res_kernel<<<dim3(RR / 128, EE / 64), 256, 0, stream>>>(
      ctx, wo_b, bo, d_query, hres, EE, EE);

  // 5) LayerNorm2 -> bf16
  ln_bf16_kernel<<<RR, 256, 0, stream>>>(hres, g2, b2, x2n);

  // 6) fused SwiGLU GEMM -> y bf16
  gemm_swiglu_kernel<<<dim3(RR / 128, FFN / 64), 256, 0, stream>>>(x2n, w12_b, b12, y);

  // 7) W3 GEMM + b3 + residual(original query) -> out fp32
  gemm_bias_res_kernel<<<dim3(RR / 128, EE / 64), 256, 0, stream>>>(
      y, w3_b, b3, d_query, out, EE, FFN);
}
